// WrappedLearnedLFrames_34059090657510
// MI455X (gfx1250) — matrix operand order repaired and checked by reading, not compiled, not run on hardware
//
#include <hip/hip_runtime.h>
#include <math.h>

#define NNODES 100000
#define NEDGES 1600000
#define DIMF   64
#define RDIM   32
#define CUT    5.0f
#define GAMMA  40.96f        // (RDIM/CUT)^2
#define WPB    4             // waves per block (wave32)

typedef __attribute__((ext_vector_type(16))) _Float16 v16h;
typedef __attribute__((ext_vector_type(8)))  float    v8f;
typedef __attribute__((ext_vector_type(8)))  unsigned v8u;
typedef __attribute__((ext_vector_type(4)))  unsigned v4u;

union Frag16 { v16h h; v8u u; };

__device__ __forceinline__ unsigned pack2(float lo, float hi) {
  union { _Float16 h[2]; unsigned u; } p;
  p.h[0] = (_Float16)lo; p.h[1] = (_Float16)hi;
  return p.u;
}

__device__ __forceinline__ float silu_f(float v) {
  return v * (1.0f / (1.0f + __expf(-v)));
}

// A-fragment (16x32 f16) from a contiguous f32 row (fallback path):
// lane<16: elems 0..7 = K0..7, 8..15 = K16..23 ; lane>=16: +8 (koff=8)
__device__ __forceinline__ v16h a_frag_row_f32(const float* __restrict__ row, int koff) {
  const float4 p0 = *(const float4*)(row + koff);
  const float4 p1 = *(const float4*)(row + koff + 4);
  const float4 p2 = *(const float4*)(row + 16 + koff);
  const float4 p3 = *(const float4*)(row + 16 + koff + 4);
  v16h a;
  a[0] = (_Float16)p0.x; a[1] = (_Float16)p0.y; a[2] = (_Float16)p0.z; a[3] = (_Float16)p0.w;
  a[4] = (_Float16)p1.x; a[5] = (_Float16)p1.y; a[6] = (_Float16)p1.z; a[7] = (_Float16)p1.w;
  a[8]  = (_Float16)p2.x; a[9]  = (_Float16)p2.y; a[10] = (_Float16)p2.z; a[11] = (_Float16)p2.w;
  a[12] = (_Float16)p3.x; a[13] = (_Float16)p3.y; a[14] = (_Float16)p3.z; a[15] = (_Float16)p3.w;
  return a;
}

// A-fragment from a pre-converted f16 row: two raw 16B loads, zero converts.
__device__ __forceinline__ v16h a_frag_row_h16(const _Float16* __restrict__ row, int koff) {
  Frag16 f;
  v4u lo = *(const v4u*)(row + koff);
  v4u hi = *(const v4u*)(row + 16 + koff);
  f.u = (v8u){lo[0], lo[1], lo[2], lo[3], hi[0], hi[1], hi[2], hi[3]};
  return f.h;
}

// Radial (Gaussian RBF) chunk computed analytically in-lane
__device__ __forceinline__ v16h radial_frag(float rn, int koff) {
  const float step = CUT / (float)(RDIM - 1);
  v16h a;
#pragma unroll
  for (int h = 0; h < 8; ++h) {
    float d0 = rn - step * (float)(koff + h);
    float d1 = rn - step * (float)(16 + koff + h);
    a[h]     = (_Float16)__expf(-GAMMA * d0 * d0);
    a[8 + h] = (_Float16)__expf(-GAMMA * d1 * d1);
  }
  return a;
}

// B-fragment from fragment-major LDS: each lane's 8 dwords are contiguous
// (32B aligned) -> two ds_load_b128 straight into the WMMA source registers.
__device__ __forceinline__ v16h b_frag_lds(const unsigned* __restrict__ wf, int frag, int lane) {
  Frag16 f;
  f.u = *(const v8u*)(wf + ((frag * 32 + lane) << 3));
  return f.h;
}

template <bool USE_H16>
__global__ __launch_bounds__(WPB * 32)
void edge_mlp_kernel(const float* __restrict__ x, const _Float16* __restrict__ xh,
                     const float* __restrict__ pos,
                     const int* __restrict__ eidx,
                     const float* __restrict__ W1, const float* __restrict__ b1,
                     const float* __restrict__ W2, const float* __restrict__ b2,
                     const float* __restrict__ W3, const float* __restrict__ b3,
                     float* __restrict__ agg, int numTiles, int totalWaves)
{
  // fragment-major B operands: [frag][lane][8 dwords]; frag = chunk*4 + ntile
  __shared__ unsigned w1f[20 * 32 * 8];   // 20 KB
  __shared__ unsigned w2f[ 8 * 32 * 8];   //  8 KB
  __shared__ float    w3s[64 * 2];
  __shared__ float    b1s[64], b2s[64], b3s[2];
  __shared__ _Float16 hbuf[WPB][16 * 64]; // per-wave activation transpose buffer

  const int tid = threadIdx.x;
  // Stage W1/W2 into WMMA B-fragment order (k-pair packed f16).
  for (int idx = tid; idx < 20 * 256; idx += WPB * 32) {
    int frag = idx >> 8;
    int l    = (idx >> 3) & 31;
    int h    = idx & 7;
    int c = frag >> 2, nt = frag & 3;
    int kp = c * 16 + ((h & 4) ? 8 : 0) + ((l & 16) ? 4 : 0) + (h & 3);   // half-pair row index
    int n  = nt * 16 + (l & 15);
    w1f[idx] = pack2(W1[(2 * kp) * 64 + n], W1[(2 * kp + 1) * 64 + n]);
  }
  for (int idx = tid; idx < 8 * 256; idx += WPB * 32) {
    int frag = idx >> 8;
    int l    = (idx >> 3) & 31;
    int h    = idx & 7;
    int kc = frag >> 2, nt = frag & 3;
    int kp = kc * 16 + ((h & 4) ? 8 : 0) + ((l & 16) ? 4 : 0) + (h & 3);
    int n  = nt * 16 + (l & 15);
    w2f[idx] = pack2(W2[(2 * kp) * 64 + n], W2[(2 * kp + 1) * 64 + n]);
  }
  if (tid < 128) w3s[tid] = W3[tid];
  if (tid < 64)  { b1s[tid] = b1[tid]; b2s[tid] = b2[tid]; }
  if (tid < 2)   b3s[tid] = b3[tid];
  __syncthreads();   // only barrier; per-wave loop below is barrier-free

  const int lane = tid & 31;
  const int wave = tid >> 5;
  const int em   = lane & 15;   // edge within 16-edge tile
  const int hi16 = lane >> 4;
  const int koff = hi16 * 8;
  _Float16* hb = hbuf[wave];

  for (int tile = blockIdx.x * WPB + wave; tile < numTiles; tile += totalWaves) {
    const int e = tile * 16 + em;
    const int j = eidx[e];            // sender
    const int i = eidx[NEDGES + e];   // receiver

    // ---- per-edge geometry (replicated on lane and lane+16) ----
    float rx = pos[3 * j + 0] - pos[3 * i + 0];
    float ry = pos[3 * j + 1] - pos[3 * i + 1];
    float rz = pos[3 * j + 2] - pos[3 * i + 2];
    float rn  = fmaxf(sqrtf(rx * rx + ry * ry + rz * rz), 1e-6f);
    float inv = 1.0f / rn;
    float hx = rx * inv, hy = ry * inv, hz = rz * inv;
    float u  = rn * (1.0f / CUT);
    float u2 = u * u, u5 = u2 * u2 * u;
    float env = 1.0f + u5 * (-21.0f + u * (35.0f - 15.0f * u));
    env = (u < 1.0f) ? env : 0.0f;

    // ---- layer 1: [16 x 160] @ [160 x 64], 5 K-chunks x 4 N-tiles ----
    v8f acc[4] = {};
#pragma unroll
    for (int c = 0; c < 5; ++c) {
      v16h a;
      if constexpr (USE_H16) {
        const _Float16* xi = xh + (size_t)i * DIMF;
        const _Float16* xj = xh + (size_t)j * DIMF;
        if      (c == 0) a = a_frag_row_h16(xi,      koff);
        else if (c == 1) a = a_frag_row_h16(xi + 32, koff);
        else if (c == 2) a = a_frag_row_h16(xj,      koff);
        else if (c == 3) a = a_frag_row_h16(xj + 32, koff);
        else             a = radial_frag(rn, koff);
      } else {
        const float* xi = x + (size_t)i * DIMF;
        const float* xj = x + (size_t)j * DIMF;
        if      (c == 0) a = a_frag_row_f32(xi,      koff);
        else if (c == 1) a = a_frag_row_f32(xi + 32, koff);
        else if (c == 2) a = a_frag_row_f32(xj,      koff);
        else if (c == 3) a = a_frag_row_f32(xj + 32, koff);
        else             a = radial_frag(rn, koff);
      }
#pragma unroll
      for (int nt = 0; nt < 4; ++nt) {
        v16h b = b_frag_lds(w1f, c * 4 + nt, lane);
        acc[nt] = __builtin_amdgcn_wmma_f32_16x16x32_f16(
            false, a, false, b, (short)0, acc[nt], false, false);
      }
    }

    // bias + SiLU, write to LDS row-major (transposes C/D layout -> A layout)
#pragma unroll
    for (int nt = 0; nt < 4; ++nt) {
      int n = nt * 16 + em;
      float bb = b1s[n];
#pragma unroll
      for (int g = 0; g < 8; ++g) {
        int M = g + hi16 * 8;
        hb[M * 64 + n] = (_Float16)silu_f(acc[nt][g] + bb);
      }
    }

    // ---- layer 2: [16 x 64] @ [64 x 64], 2 K-chunks x 4 N-tiles ----
    v8f acc2[4] = {};
#pragma unroll
    for (int kc = 0; kc < 2; ++kc) {
      v16h a = a_frag_row_h16(hb + em * 64 + kc * 32, koff);
#pragma unroll
      for (int nt = 0; nt < 4; ++nt) {
        v16h b = b_frag_lds(w2f, kc * 4 + nt, lane);
        acc2[nt] = __builtin_amdgcn_wmma_f32_16x16x32_f16(
            false, a, false, b, (short)0, acc2[nt], false, false);
      }
    }

#pragma unroll
    for (int nt = 0; nt < 4; ++nt) {
      int n = nt * 16 + em;
      float bb = b2s[n];
#pragma unroll
      for (int g = 0; g < 8; ++g) {
        int M = g + hi16 * 8;
        hb[M * 64 + n] = (_Float16)silu_f(acc2[nt][g] + bb);
      }
    }

    // ---- layer 3: [16 x 64] @ [64 x 2] via VALU (lane = edge x pred) ----
    float dot = b3s[hi16];
#pragma unroll 8
    for (int k = 0; k < 64; ++k)
      dot += (float)hb[em * 64 + k] * w3s[k * 2 + hi16];

    // broadcast both predictions to the lane pair via wave32 shuffles
    float m0 = __shfl(dot, em,      32);
    float m1 = __shfl(dot, em + 16, 32);

    // ---- message + scatter-add: split 9 atomics across the lane pair ----
    float* dst = agg + (size_t)i * 9;
    if (lane < 16) {
      atomicAdd(dst + 0, m0 * hx * env);
      atomicAdd(dst + 1, m0 * hy * env);
      atomicAdd(dst + 2, m0 * hz * env);
      atomicAdd(dst + 3, m1 * hx * env);
      atomicAdd(dst + 4, m1 * hy * env);
    } else {
      atomicAdd(dst + 5, m1 * hz * env);
      atomicAdd(dst + 6, hx * env);
      atomicAdd(dst + 7, hy * env);
      atomicAdd(dst + 8, hz * env);
    }
  }
}

__global__ void zero_kernel(float* p, int n) {
  int t = blockIdx.x * blockDim.x + threadIdx.x;
  if (t < n) p[t] = 0.0f;
}

// Convert x (f32) into a packed f16 mirror; 8 elements per thread.
__global__ void xcvt_kernel(const float* __restrict__ src, _Float16* __restrict__ dst, int n8) {
  int t = blockIdx.x * blockDim.x + threadIdx.x;
  if (t >= n8) return;
  const float4 p0 = *(const float4*)(src + t * 8);
  const float4 p1 = *(const float4*)(src + t * 8 + 4);
  v4u o;
  o[0] = pack2(p0.x, p0.y);
  o[1] = pack2(p0.z, p0.w);
  o[2] = pack2(p1.x, p1.y);
  o[3] = pack2(p1.z, p1.w);
  *(v4u*)(dst + t * 8) = o;
}

__device__ __forceinline__ void normalize3(float& a, float& b, float& c) {
  float n = sqrtf(a * a + b * b + c * c);
  float inv = 1.0f / fmaxf(n, 1e-6f);
  a *= inv; b *= inv; c *= inv;
}

// NOTE: no __restrict__ here — agg may alias out (in-place fallback);
// each thread reads its 9 inputs fully before writing its 9 outputs.
__global__ void frames_kernel(const float* agg, float* out, int n) {
  int t = blockIdx.x * blockDim.x + threadIdx.x;
  if (t >= n) return;
  const float* a = agg + (size_t)t * 9;
  float v0x = a[0], v0y = a[1], v0z = a[2];
  float v1x = a[3], v1y = a[4], v1z = a[5];
  float v2x = a[6], v2y = a[7], v2z = a[8];

  float e1x = v0x, e1y = v0y, e1z = v0z;
  normalize3(e1x, e1y, e1z);
  float d1 = v1x * e1x + v1y * e1y + v1z * e1z;
  float e2x = v1x - d1 * e1x, e2y = v1y - d1 * e1y, e2z = v1z - d1 * e1z;
  normalize3(e2x, e2y, e2z);
  float d2a = v2x * e1x + v2y * e1y + v2z * e1z;
  float d2b = v2x * e2x + v2y * e2y + v2z * e2z;
  float e3x = v2x - d2a * e1x - d2b * e2x;
  float e3y = v2y - d2a * e1y - d2b * e2y;
  float e3z = v2z - d2a * e1z - d2b * e2z;
  normalize3(e3x, e3y, e3z);

  float* o = out + (size_t)t * 9;
  o[0] = e1x; o[1] = e1y; o[2] = e1z;
  o[3] = e2x; o[4] = e2y; o[5] = e2z;
  o[6] = e3x; o[7] = e3y; o[8] = e3z;
}

__global__ void copyx_kernel(const float4* __restrict__ src, float4* __restrict__ dst, int n) {
  int t = blockIdx.x * blockDim.x + threadIdx.x;
  if (t < n) dst[t] = src[t];
}

extern "C" void kernel_launch(void* const* d_in, const int* in_sizes, int n_in,
                              void* d_out, int out_size, void* d_ws, size_t ws_size,
                              hipStream_t stream) {
  (void)in_sizes; (void)n_in; (void)out_size;
  const float* x   = (const float*)d_in[0];
  const float* pos = (const float*)d_in[1];
  const int*   ei  = (const int*)d_in[2];
  // d_in[3] = batch (unused: single graph)
  const float* W1  = (const float*)d_in[4];
  const float* b1  = (const float*)d_in[5];
  const float* W2  = (const float*)d_in[6];
  const float* b2  = (const float*)d_in[7];
  const float* W3  = (const float*)d_in[8];
  const float* b3  = (const float*)d_in[9];

  const int    aggElems = NNODES * 9;
  const size_t aggBytes = (size_t)aggElems * sizeof(float);          // 3.6 MB
  const size_t xhBytes  = (size_t)NNODES * DIMF * sizeof(_Float16);  // 12.8 MB
  const bool haveAgg = (ws_size >= aggBytes);
  const bool haveXh  = (ws_size >= aggBytes + xhBytes);

  float*     agg = haveAgg ? (float*)d_ws : (float*)d_out;
  _Float16*  xhp = (_Float16*)((char*)d_ws + aggBytes);

  zero_kernel<<<(aggElems + 255) / 256, 256, 0, stream>>>(agg, aggElems);

  const int numTiles = NEDGES / 16;   // E is an exact multiple of 16
  const int blocks = 1200;
  if (haveXh) {
    const int n8 = NNODES * DIMF / 8;
    xcvt_kernel<<<(n8 + 255) / 256, 256, 0, stream>>>(x, xhp, n8);
    edge_mlp_kernel<true><<<blocks, WPB * 32, 0, stream>>>(
        x, xhp, pos, ei, W1, b1, W2, b2, W3, b3, agg, numTiles, blocks * WPB);
  } else {
    edge_mlp_kernel<false><<<blocks, WPB * 32, 0, stream>>>(
        x, (const _Float16*)nullptr, pos, ei, W1, b1, W2, b2, W3, b3,
        agg, numTiles, blocks * WPB);
  }

  frames_kernel<<<(NNODES + 255) / 256, 256, 0, stream>>>(agg, (float*)d_out, NNODES);

  const int n4 = NNODES * DIMF / 4;
  copyx_kernel<<<(n4 + 255) / 256, 256, 0, stream>>>(
      (const float4*)x, ((float4*)d_out) + (NNODES * 9) / 4, n4);
}
